// MultiHeadAttention_53798760350257
// MI455X (gfx1250) — matrix operand order repaired and checked by reading, not compiled, not run on hardware
//
#include <hip/hip_runtime.h>

typedef __attribute__((ext_vector_type(16))) __bf16 v16bf;
typedef __attribute__((ext_vector_type(8)))  __bf16 v8bf;
typedef __attribute__((ext_vector_type(8)))  float  v8f;
typedef __attribute__((ext_vector_type(4)))  unsigned int v4u;
typedef __attribute__((ext_vector_type(8)))  unsigned int v8u;
typedef unsigned short u16;
typedef unsigned int   u32;
typedef unsigned long long u64;

// ---------------------------------------------------------------- helpers ---
__device__ __forceinline__ u16 f2bf(float f) {
  u32 u = __float_as_uint(f);
  u32 r = u + 0x7FFFu + ((u >> 16) & 1u);   // round-to-nearest-even
  return (u16)(r >> 16);
}

// Raw workgroup split-barrier: no conservative vmem fence (LDS producer/consumer
// ordering here is guaranteed by s_wait_tensorcnt before signal + dscnt drains).
__device__ __forceinline__ void wg_barrier() {
  asm volatile("s_barrier_signal -1\n\ts_barrier_wait -1" ::: "memory");
}

// A-fragment (16x32 bf16, M=lane&15): per ISA swizzle, lane half h holds
// K = 8h..8h+7 (elems 0-7) and K = 16+8h..16+8h+7 (elems 8-15).
__device__ __forceinline__ v16bf load_a_frag(const u16* rowp, int hf) {
  const __bf16* p = (const __bf16*)rowp + hf * 8;
  v8bf lo = *(const v8bf*)(p);
  v8bf hi = *(const v8bf*)(p + 16);
  v16bf r;
#pragma unroll
  for (int i = 0; i < 8; ++i) { r[i] = lo[i]; r[8 + i] = hi[i]; }
  return r;
}

// B-fragment (32x16 bf16, N=lane&15): lane half h holds K = 8h..8h+7 (elems 0-7)
// and K = 16+8h..16+8h+7 (elems 8-15), matching the A-fragment.
__device__ __forceinline__ v16bf load_b_frag(const u16* rowp, int hf) {
  const __bf16* p = (const __bf16*)rowp + hf * 8;
  v8bf lo = *(const v8bf*)(p);
  v8bf hi = *(const v8bf*)(p + 16);
  v16bf r;
#pragma unroll
  for (int i = 0; i < 8; ++i) { r[i] = lo[i]; r[8 + i] = hi[i]; }
  return r;
}

__device__ __forceinline__ v8f wmma_bf16(v16bf a, v16bf b, v8f c) {
  return __builtin_amdgcn_wmma_f32_16x16x32_bf16(false, a, false, b, (short)0, c,
                                                 false, false);
}

__device__ __forceinline__ u32 lds_off(const void* p) {
  return (u32)(unsigned long long)p;   // flat LDS address carries offset in [31:0]
}

// TDM: DMA a 2D bf16 tile (tile_d0 x tile_d1, row stride stride0 elems) from
// global into LDS at lds_byte. Descriptors are wave-uniform -> SGPR groups.
__device__ __forceinline__ void tdm_load_2d(const u16* gsrc, u32 lds_byte,
                                            u32 tile_d0, u32 tile_d1,
                                            u32 tensor_d0, u32 tensor_d1,
                                            u32 stride0) {
  u64 ga = (u64)(unsigned long long)gsrc;
  v4u g0;
  g0[0] = 1u;                                           // count=1, user mode
  g0[1] = lds_byte;                                     // lds_addr
  g0[2] = (u32)ga;                                      // global_addr[31:0]
  g0[3] = (u32)((ga >> 32) & 0x01FFFFFFu) | (2u << 30); // addr[56:32] | type=2
  v8u g1;
  g1[0] = (1u << 16);                                   // data_size=1 (2 bytes)
  g1[1] = (tensor_d0 & 0xFFFFu) << 16;                  // tensor_dim0[15:0]
  g1[2] = (tensor_d0 >> 16) | ((tensor_d1 & 0xFFFFu) << 16);
  g1[3] = (tensor_d1 >> 16) | (tile_d0 << 16);          // tile_dim0
  g1[4] = tile_d1;                                      // tile_dim1 (tile_dim2=0)
  g1[5] = stride0;                                      // tensor_dim0_stride[31:0]
  g1[6] = 0;
  g1[7] = 0;
  asm volatile("tensor_load_to_lds %0, %1" :: "s"(g0), "s"(g1) : "memory");
}

// ------------------------------------------------------------ fp32 -> bf16 ---
__global__ __launch_bounds__(256) void cvt_f32_bf16(const float* __restrict__ in,
                                                    u16* __restrict__ out) {
  int i = (blockIdx.x * 256 + threadIdx.x) * 4;
  float4 v = *(const float4*)(in + i);
  ushort4 o = make_ushort4(f2bf(v.x), f2bf(v.y), f2bf(v.z), f2bf(v.w));
  *(ushort4*)(out + i) = o;
}

// ------------------------------------------------------ QKV projection GEMM ---
// C[m,n] = sum_k x[m,k] * W[n,k] + b[n]; block = 128(M) x 64(N), wave = 16x64.
// Shared 64x32 W-tile staged by TDM into double-buffered LDS; A pipelined in regs.
__global__ __launch_bounds__(256)
void qkv_proj(const u16* __restrict__ xbf, const u16* __restrict__ wqkv,
              const float* __restrict__ bq, const float* __restrict__ bk,
              const float* __restrict__ bv, u16* __restrict__ Qb,
              u16* __restrict__ Kb, u16* __restrict__ Vtb) {
  __shared__ u16 bsm[2][64 * 32];         // 2 x 4KB W tiles
  const int lane = threadIdx.x & 31;
  const int wave = threadIdx.x >> 5;
  const int row  = lane & 15;
  const int hf   = lane >> 4;
  const int z    = blockIdx.z;
  const int n0   = blockIdx.x * 64;
  const int m0   = blockIdx.y * 128 + wave * 16;

  const u16*   W    = wqkv + (size_t)z * (1024 * 1024);
  const float* bias = (z == 0) ? bq : (z == 1) ? bk : bv;
  const u16*   arow = xbf + (size_t)(m0 + row) * 1024;
  const u16*   Wt   = W + (size_t)n0 * 1024;

  if (wave == 0)
    tdm_load_2d(Wt, lds_off(&bsm[0][0]), 32, 64, 1024, 1024, 1024);

  v16bf a = load_a_frag(arow, hf);        // prologue A (k = 0)
  v8f acc[4] = {};
  int buf = 0;
  for (int k = 0; k < 1024; k += 32) {
    wg_barrier();                         // buf^1 reads (prev iter) drained
    if (wave == 0) {
      if (k + 32 < 1024) {
        tdm_load_2d(Wt + (k + 32), lds_off(&bsm[buf ^ 1][0]), 32, 64, 1024, 1024, 1024);
        __builtin_amdgcn_s_wait_tensorcnt(1);   // current tile done
      } else {
        __builtin_amdgcn_s_wait_tensorcnt(0);
      }
    }
    wg_barrier();                         // tile in bsm[buf] visible to all
    v16bf b0 = load_b_frag(&bsm[buf][(0 * 16 + row) * 32], hf);
    v16bf b1 = load_b_frag(&bsm[buf][(1 * 16 + row) * 32], hf);
    v16bf b2 = load_b_frag(&bsm[buf][(2 * 16 + row) * 32], hf);
    v16bf b3 = load_b_frag(&bsm[buf][(3 * 16 + row) * 32], hf);
    v16bf an = a;
    if (k + 32 < 1024) {                  // pipeline next A under the WMMAs
      __builtin_prefetch(arow + k + 288, 0, 3);
      an = load_a_frag(arow + k + 32, hf);
    }
    acc[0] = wmma_bf16(a, b0, acc[0]);
    acc[1] = wmma_bf16(a, b1, acc[1]);
    acc[2] = wmma_bf16(a, b2, acc[2]);
    acc[3] = wmma_bf16(a, b3, acc[3]);
    a = an;
    buf ^= 1;
  }

  const int bb = m0 >> 11;
  const int sb = m0 & 2047;
#pragma unroll
  for (int t = 0; t < 4; ++t) {
    int n = n0 + t * 16 + row;
    float bval = bias[n];
    int h = n >> 6, d = n & 63;
#pragma unroll
    for (int r = 0; r < 8; ++r) {
      int s = sb + r + 8 * hf;
      u16 v = f2bf(acc[t][r] + bval);
      if (z == 2)
        Vtb[((size_t)(bb * 16 + h) * 64 + d) * 2048 + s] = v;
      else {
        u16* dst = z ? Kb : Qb;
        dst[((size_t)(bb * 16 + h) * 2048 + s) * 64 + d] = v;
      }
    }
  }
}

// ---------------------------------------------------- flash-style attention ---
// One wave owns 16 query rows; K/V^T key tiles (shared by all 8 waves) are TDM-
// staged into double-buffered LDS; online softmax; P restaged via per-wave LDS.
__global__ __launch_bounds__(256)
void attn(const u16* __restrict__ Qb, const u16* __restrict__ Kb,
          const u16* __restrict__ Vtb, u16* __restrict__ ctx) {
  __shared__ u16 plds[8 * 16 * 32];       // per-wave 16x32 bf16 P tile
  __shared__ u16 ksm[2][32 * 64];         // 2 x 4KB K tiles  [key][d]
  __shared__ u16 vsm[2][64 * 32];         // 2 x 4KB V^T tiles [d][key]
  const int lane = threadIdx.x & 31;
  const int wave = threadIdx.x >> 5;
  const int row  = lane & 15;
  const int hf   = lane >> 4;
  const int bh   = blockIdx.y;
  const int q0   = blockIdx.x * 128 + wave * 16;

  const u16* Qrow = Qb + ((size_t)bh * 2048 + q0 + row) * 64;
  v16bf qa0 = load_a_frag(Qrow, hf);      // d = 0..31
  v16bf qa1 = load_a_frag(Qrow + 32, hf); // d = 32..63

  const u16* Kbase = Kb  + (size_t)bh * 2048 * 64;
  const u16* Vbase = Vtb + (size_t)bh * 64 * 2048;
  u16* pl = &plds[wave * (16 * 32)];

  if (wave == 0) {
    tdm_load_2d(Kbase, lds_off(&ksm[0][0]), 64, 32, 64, 2048, 64);
    tdm_load_2d(Vbase, lds_off(&vsm[0][0]), 32, 64, 2048, 64, 2048);
  }

  v8f o[4] = {};
  float mrow[8], lrow[8];
#pragma unroll
  for (int r = 0; r < 8; ++r) { mrow[r] = -3.0e38f; lrow[r] = 0.f; }

  int buf = 0;
  for (int j = 0; j < 2048; j += 32) {
    wg_barrier();                         // prev reads of buf^1 drained
    if (wave == 0) {
      if (j + 32 < 2048) {
        tdm_load_2d(Kbase + (size_t)(j + 32) * 64, lds_off(&ksm[buf ^ 1][0]),
                    64, 32, 64, 2048, 64);
        tdm_load_2d(Vbase + (j + 32), lds_off(&vsm[buf ^ 1][0]),
                    32, 64, 2048, 64, 2048);
        __builtin_amdgcn_s_wait_tensorcnt(2);   // current K+V tiles done
      } else {
        __builtin_amdgcn_s_wait_tensorcnt(0);
      }
    }
    wg_barrier();                         // tiles visible to all waves

    // batched K-fragment loads, then 4 back-to-back WMMAs
    const u16* kl = &ksm[buf][row * 64];
    v16bf kb0 = load_b_frag(kl, hf);
    v16bf kb1 = load_b_frag(kl + 32, hf);
    v16bf kb2 = load_b_frag(kl + 16 * 64, hf);
    v16bf kb3 = load_b_frag(kl + 16 * 64 + 32, hf);
    // prefetch V fragments early; latency hides under the softmax VALU block
    v16bf vb0 = load_b_frag(&vsm[buf][(0 * 16 + row) * 32], hf);
    v16bf vb1 = load_b_frag(&vsm[buf][(1 * 16 + row) * 32], hf);
    v16bf vb2 = load_b_frag(&vsm[buf][(2 * 16 + row) * 32], hf);
    v16bf vb3 = load_b_frag(&vsm[buf][(3 * 16 + row) * 32], hf);
    v8f s0 = {}, s1 = {};
    s0 = wmma_bf16(qa0, kb0, s0);
    s0 = wmma_bf16(qa1, kb1, s0);
    s1 = wmma_bf16(qa0, kb2, s1);
    s1 = wmma_bf16(qa1, kb3, s1);

#pragma unroll
    for (int r = 0; r < 8; ++r) {
      float a0 = s0[r] * 0.125f;          // 1/sqrt(64)
      float a1 = s1[r] * 0.125f;
      float mx = fmaxf(a0, a1);           // row reduction within 16-lane half
      mx = fmaxf(mx, __shfl_xor(mx, 1));
      mx = fmaxf(mx, __shfl_xor(mx, 2));
      mx = fmaxf(mx, __shfl_xor(mx, 4));
      mx = fmaxf(mx, __shfl_xor(mx, 8));
      float mn = fmaxf(mrow[r], mx);
      float corr = __expf(mrow[r] - mn);
      mrow[r] = mn;
      float p0 = __expf(a0 - mn);
      float p1 = __expf(a1 - mn);
      float rs = p0 + p1;
      rs += __shfl_xor(rs, 1);
      rs += __shfl_xor(rs, 2);
      rs += __shfl_xor(rs, 4);
      rs += __shfl_xor(rs, 8);
      lrow[r] = lrow[r] * corr + rs;
#pragma unroll
      for (int t = 0; t < 4; ++t) o[t][r] = o[t][r] * corr;
      int pr = r + 8 * hf;
      pl[pr * 32 + row]      = f2bf(p0);
      pl[pr * 32 + row + 16] = f2bf(p1);
    }

    v16bf pa = load_a_frag(pl + row * 32, hf);  // P as A-fragment (same wave)
    o[0] = wmma_bf16(pa, vb0, o[0]);
    o[1] = wmma_bf16(pa, vb1, o[1]);
    o[2] = wmma_bf16(pa, vb2, o[2]);
    o[3] = wmma_bf16(pa, vb3, o[3]);
    buf ^= 1;
  }

  const int bb = bh >> 4, h = bh & 15;
#pragma unroll
  for (int t = 0; t < 4; ++t)
#pragma unroll
    for (int r = 0; r < 8; ++r) {
      int q = q0 + r + 8 * hf;
      int d = t * 16 + row;
      float val = o[t][r] / lrow[r];
      ctx[(size_t)(bb * 2048 + q) * 1024 + h * 64 + d] = f2bf(val);
    }
}

// -------------------------------------------------------- output projection ---
__global__ __launch_bounds__(256)
void out_proj(const u16* __restrict__ ctxb, const u16* __restrict__ wo,
              const float* __restrict__ bo, float* __restrict__ out) {
  __shared__ u16 bsm[2][64 * 32];
  const int lane = threadIdx.x & 31;
  const int wave = threadIdx.x >> 5;
  const int row  = lane & 15;
  const int hf   = lane >> 4;
  const int n0   = blockIdx.x * 64;
  const int m0   = blockIdx.y * 128 + wave * 16;

  const u16* arow = ctxb + (size_t)(m0 + row) * 1024;
  const u16* Wt   = wo + (size_t)n0 * 1024;

  if (wave == 0)
    tdm_load_2d(Wt, lds_off(&bsm[0][0]), 32, 64, 1024, 1024, 1024);

  v16bf a = load_a_frag(arow, hf);
  v8f acc[4] = {};
  int buf = 0;
  for (int k = 0; k < 1024; k += 32) {
    wg_barrier();
    if (wave == 0) {
      if (k + 32 < 1024) {
        tdm_load_2d(Wt + (k + 32), lds_off(&bsm[buf ^ 1][0]), 32, 64, 1024, 1024, 1024);
        __builtin_amdgcn_s_wait_tensorcnt(1);
      } else {
        __builtin_amdgcn_s_wait_tensorcnt(0);
      }
    }
    wg_barrier();
    v16bf b0 = load_b_frag(&bsm[buf][(0 * 16 + row) * 32], hf);
    v16bf b1 = load_b_frag(&bsm[buf][(1 * 16 + row) * 32], hf);
    v16bf b2 = load_b_frag(&bsm[buf][(2 * 16 + row) * 32], hf);
    v16bf b3 = load_b_frag(&bsm[buf][(3 * 16 + row) * 32], hf);
    v16bf an = a;
    if (k + 32 < 1024) {
      __builtin_prefetch(arow + k + 288, 0, 3);
      an = load_a_frag(arow + k + 32, hf);
    }
    acc[0] = wmma_bf16(a, b0, acc[0]);
    acc[1] = wmma_bf16(a, b1, acc[1]);
    acc[2] = wmma_bf16(a, b2, acc[2]);
    acc[3] = wmma_bf16(a, b3, acc[3]);
    a = an;
    buf ^= 1;
  }

#pragma unroll
  for (int t = 0; t < 4; ++t) {
    int n = n0 + t * 16 + row;
    float bval = bo[n];
#pragma unroll
    for (int r = 0; r < 8; ++r) {
      int m = m0 + r + 8 * hf;
      out[(size_t)m * 1024 + n] = acc[t][r] + bval;
    }
  }
}

// ----------------------------------------------------------------- launcher ---
extern "C" void kernel_launch(void* const* d_in, const int* in_sizes, int n_in,
                              void* d_out, int out_size, void* d_ws, size_t ws_size,
                              hipStream_t stream) {
  const float* x  = (const float*)d_in[0];
  const float* Wq = (const float*)d_in[1];
  const float* bq = (const float*)d_in[2];
  const float* Wk = (const float*)d_in[3];
  const float* bk = (const float*)d_in[4];
  const float* Wv = (const float*)d_in[5];
  const float* bv = (const float*)d_in[6];
  const float* Wo = (const float*)d_in[7];
  const float* bo = (const float*)d_in[8];
  float* out = (float*)d_out;

  const size_t XN = (size_t)8192 * 1024;
  const size_t WN = (size_t)1024 * 1024;

  u16* ws   = (u16*)d_ws;
  u16* xbf  = ws;            // 16 MB
  u16* wq   = xbf + XN;      // wq|wk|wv contiguous for z-indexing
  u16* wk   = wq + WN;
  u16* wv   = wk + WN;
  u16* wo   = wv + WN;
  u16* Qb   = wo + WN;       // [b,h,s,d]
  u16* Kbuf = Qb + XN;       // [b,h,s,d]
  u16* Vtb  = Kbuf + XN;     // [b,h,d,s]
  u16* ctxb = Vtb + XN;      // [b,s,h*d]
  (void)in_sizes; (void)n_in; (void)out_size; (void)ws_size;

  cvt_f32_bf16<<<8192, 256, 0, stream>>>(x, xbf);
  cvt_f32_bf16<<<1024, 256, 0, stream>>>(Wq, wq);
  cvt_f32_bf16<<<1024, 256, 0, stream>>>(Wk, wk);
  cvt_f32_bf16<<<1024, 256, 0, stream>>>(Wv, wv);
  cvt_f32_bf16<<<1024, 256, 0, stream>>>(Wo, wo);

  dim3 gq(16, 64, 3);
  qkv_proj<<<gq, 256, 0, stream>>>(xbf, wq, bq, bk, bv, Qb, Kbuf, Vtb);

  dim3 ga(16, 64);
  attn<<<ga, 256, 0, stream>>>(Qb, Kbuf, Vtb, ctxb);

  dim3 go(16, 64);
  out_proj<<<go, 256, 0, stream>>>(ctxb, wo, bo, out);
}
